// CloudCrop_33397665693880
// MI455X (gfx1250) — matrix operand order repaired and checked by reading, not compiled, not run on hardware
//
#include <hip/hip_runtime.h>

// ---------------- problem constants ----------------
#define B_      8
#define N_      1024
#define C_      256
#define NS_     32
#define CIN_    259          // 3 + C
#define KP1     288          // CIN padded to multiple of 32
#define KP2     256
#define O_      256
#define COLS    (B_*N_*NS_)  // 262144
#define NTILE   64           // columns per GEMM block
#define RADIUS_ 0.05f
#define HMIN_   (-0.02f)
#define HMAX_   0.04f
#define EPS_    1e-5f

typedef _Float16 half_t;
typedef __attribute__((ext_vector_type(16))) _Float16 v16h;
typedef __attribute__((ext_vector_type(8)))  _Float16 v8h;
typedef __attribute__((ext_vector_type(8)))  float    v8f;

__device__ __forceinline__ v16h cat8(v8h lo, v8h hi) {
    return __builtin_shufflevector(lo, hi, 0,1,2,3,4,5,6,7,8,9,10,11,12,13,14,15);
}

// ---------------- K0a: zero stat accumulators ----------------
__global__ void zero_stats_kernel(float* stats) {
    int t = blockIdx.x * blockDim.x + threadIdx.x;
    if (t < 2048) stats[t] = 0.0f;
}

// ---------------- K0b: feats (B,C,N) f32 -> featsT (B,N,C) f16 ----------------
__global__ __launch_bounds__(256)
void prep_feats_kernel(const float* __restrict__ feats, half_t* __restrict__ featsT) {
    int t = blockIdx.x * 256 + threadIdx.x;            // t = (b*N+n)*C + c
    if (t >= B_ * N_ * C_) return;
    int b   = t >> 18;                                 // N*C = 2^18
    int rem = t & (N_ * C_ - 1);
    int n   = rem >> 8;
    int c   = rem & (C_ - 1);
    featsT[t] = (half_t)feats[((size_t)(b * C_ + c)) * N_ + n];
}

// ---------------- K0c: W1 (256,259) f32 -> K-permuted padded f16 [256][288] ----------------
// Permutation: k in [0,256) <- W1 col 3+k (features); k in [256,259) <- W1 col k-256 (xyz); else 0.
__global__ __launch_bounds__(256)
void prep_w1_kernel(const float* __restrict__ W1, half_t* __restrict__ W1h) {
    int t = blockIdx.x * 256 + threadIdx.x;
    if (t >= O_ * KP1) return;
    int o = t / KP1, k = t - o * KP1;
    float v = 0.0f;
    if (k < 256)          v = W1[(size_t)o * CIN_ + 3 + k];
    else if (k < CIN_)    v = W1[(size_t)o * CIN_ + (k - 256)];
    W1h[t] = (half_t)v;
}

// ---------------- K0d: W2 (256,256) f32 -> f16 row-major ----------------
__global__ __launch_bounds__(256)
void prep_w2_kernel(const float* __restrict__ W2, half_t* __restrict__ W2h) {
    int t = blockIdx.x * 256 + threadIdx.x;
    if (t >= O_ * KP2) return;
    W2h[t] = (half_t)W2[t];
}

// ---------------- K1: cylinder query (one wave32 per center) ----------------
__global__ __launch_bounds__(256)
void cyl_query_kernel(const float* __restrict__ xyz, const float* __restrict__ rot,
                      int* __restrict__ idxOut) {
    int lane   = threadIdx.x & 31;
    int center = blockIdx.x * 8 + (threadIdx.x >> 5);  // = b*N + p
    int b = center >> 10;

    const float* cen = xyz + (size_t)center * 3;
    const float* R   = rot + (size_t)center * 9;
    float cx = cen[0], cy = cen[1], cz = cen[2];
    float r0 = R[0], r1 = R[1], r2 = R[2];
    float r3 = R[3], r4 = R[4], r5 = R[5];
    float r6 = R[6], r7 = R[7], r8 = R[8];

    idxOut[center * NS_ + lane] = 0;                   // default slot (same-wave stores stay ordered)

    int cnt = 0;
    for (int chunk = 0; chunk < N_ / 32 && cnt < NS_; ++chunk) {
        int i = chunk * 32 + lane;
        const float* pt = xyz + ((size_t)(b * N_ + i)) * 3;
        float dx = pt[0] - cx, dy = pt[1] - cy, dz = pt[2] - cz;
        float l0 = dx * r0 + dy * r3 + dz * r6;        // local_k = sum_m diff_m * rot[m][k]
        float l1 = dx * r1 + dy * r4 + dz * r7;
        float l2 = dx * r2 + dy * r5 + dz * r8;
        bool m = (l1 * l1 + l2 * l2 < RADIUS_ * RADIUS_) && (l0 > HMIN_) && (l0 < HMAX_);
        unsigned bal = (unsigned)__builtin_amdgcn_ballot_w32(m);
        int pos = cnt + __popc(bal & ((1u << lane) - 1u));
        if (m && pos < NS_) idxOut[center * NS_ + pos] = i;
        cnt += __popc(bal);
    }
}

// ---------------- K2: GEMM1, 256(M) x 64(N) tile per block + BN1 stats ----------------
__global__ __launch_bounds__(256)
void gemm1_kernel(const int* __restrict__ idx, const float* __restrict__ xyz,
                  const float* __restrict__ rot, const half_t* __restrict__ featsT,
                  const half_t* __restrict__ W1h, half_t* __restrict__ y1,
                  float* __restrict__ gsum, float* __restrict__ gsq) {
    __shared__ half_t Bs[NTILE * KP1];                 // 36 KB
    __shared__ float  s_sum[O_], s_sq[O_];

    int tid = threadIdx.x;
    int colBase = blockIdx.x * NTILE;
    s_sum[tid] = 0.0f; s_sq[tid] = 0.0f;

    // ---- stage 64 columns x 288 deep: 4 threads per column ----
    int col = tid >> 2, r = tid & 3;
    int gcol = colBase + col;
    int b   = gcol >> 15;                              // N*NS = 2^15
    int rem = gcol & (N_ * NS_ - 1);
    int p   = rem >> 5;
    int i   = idx[gcol];

    // features at k=0..255 (aligned v8h copies, 64 halves per thread)
    {
        const half_t* src = featsT + ((size_t)(b * N_ + i)) * C_ + r * 64;
        half_t*       dst = Bs + (size_t)col * KP1 + r * 64;
        #pragma unroll
        for (int j = 0; j < 8; ++j)
            *(v8h*)(dst + j * 8) = *(const v8h*)(src + j * 8);
    }
    if (r == 0) {                                      // rotated, radius-normalized xyz at k=256..258
        const float* cen = xyz + ((size_t)(b * N_ + p)) * 3;
        const float* pt  = xyz + ((size_t)(b * N_ + i)) * 3;
        const float* R   = rot + ((size_t)(b * N_ + p)) * 9;
        float d0 = pt[0] - cen[0], d1 = pt[1] - cen[1], d2 = pt[2] - cen[2];
        #pragma unroll
        for (int k = 0; k < 3; ++k) {
            float v = d0 * R[k] + d1 * R[3 + k] + d2 * R[6 + k];
            Bs[(size_t)col * KP1 + 256 + k] = (half_t)(v * (1.0f / RADIUS_));
        }
        #pragma unroll
        for (int k = 259; k < 264; ++k) Bs[(size_t)col * KP1 + k] = (half_t)0.0f;
    }
    if (r == 3) {                                      // zero pad k=264..287 (16B aligned)
        v8h z = {};
        *(v8h*)(Bs + (size_t)col * KP1 + 264) = z;
        *(v8h*)(Bs + (size_t)col * KP1 + 272) = z;
        *(v8h*)(Bs + (size_t)col * KP1 + 280) = z;
    }
    __syncthreads();

    // ---- WMMA: 8 waves, each 2 M-tiles x 4 N-tiles (8 accumulators) ----
    int wave = tid >> 5, lane = tid & 31;
    int m0 = wave * 32;
    int n  = lane & 15, hi = lane >> 4;
    v8f acc[8];
    #pragma unroll
    for (int q = 0; q < 8; ++q) acc[q] = (v8f){};
    const half_t* arow0 = W1h + (size_t)(m0 + n) * KP1 + hi * 8;   // A: lane=row, split K
    const half_t* arow1 = arow0 + (size_t)16 * KP1;

    #pragma unroll
    for (int k0 = 0; k0 < KP1; k0 += 32) {
        v16h a0 = cat8(*(const v8h*)(arow0 + k0), *(const v8h*)(arow0 + k0 + 16));
        v16h a1 = cat8(*(const v8h*)(arow1 + k0), *(const v8h*)(arow1 + k0 + 16));
        #pragma unroll
        for (int nt = 0; nt < 4; ++nt) {
            const half_t* bcol = Bs + (size_t)(nt * 16 + n) * KP1 + hi * 8 + k0;
            v16h bb = cat8(*(const v8h*)(bcol), *(const v8h*)(bcol + 16));
            acc[nt]     = __builtin_amdgcn_wmma_f32_16x16x32_f16(false, a0, false, bb, (short)0, acc[nt],     false, false);
            acc[4 + nt] = __builtin_amdgcn_wmma_f32_16x16x32_f16(false, a1, false, bb, (short)0, acc[4 + nt], false, false);
        }
    }

    // ---- write y1 (f16, [col][o]) + per-channel sum/sumsq ----
    #pragma unroll
    for (int nt = 0; nt < 4; ++nt) {
        v8h pk0, pk1;
        #pragma unroll
        for (int rr = 0; rr < 8; ++rr) {               // D: VGPR rr -> M = rr + hi*8
            float v0 = acc[nt][rr], v1 = acc[4 + nt][rr];
            int o0 = m0 + hi * 8 + rr, o1 = o0 + 16;
            atomicAdd(&s_sum[o0], v0); atomicAdd(&s_sq[o0], v0 * v0);
            atomicAdd(&s_sum[o1], v1); atomicAdd(&s_sq[o1], v1 * v1);
            pk0[rr] = (half_t)v0; pk1[rr] = (half_t)v1;
        }
        half_t* yrow = y1 + (size_t)(colBase + nt * 16 + n) * O_;
        *(v8h*)(yrow + m0 + hi * 8)      = pk0;
        *(v8h*)(yrow + m0 + 16 + hi * 8) = pk1;
    }

    __syncthreads();
    atomicAdd(&gsum[tid], s_sum[tid]);
    atomicAdd(&gsq[tid],  s_sq[tid]);
}

// ---------------- K3/K5: finalize BN stats ----------------
__global__ void finalize_stats_kernel(float* stats) {  // [0..255]=sum [256..511]=sumsq -> [512]=mu [768]=rsig
    int o = threadIdx.x;
    float cnt = (float)COLS;
    float mu  = stats[o] / cnt;
    float var = stats[256 + o] / cnt - mu * mu;
    stats[512 + o] = mu;
    stats[768 + o] = rsqrtf(var + EPS_);
}

// ---------------- K4: BN1+ReLU -> GEMM2 (256x64 tile) + BN2 stats ----------------
__global__ __launch_bounds__(256)
void gemm2_kernel(const half_t* __restrict__ y1, const float* __restrict__ mu1,
                  const float* __restrict__ rs1, const float* __restrict__ g1,
                  const float* __restrict__ b1, const half_t* __restrict__ W2h,
                  half_t* __restrict__ y2, float* __restrict__ gsum, float* __restrict__ gsq) {
    __shared__ half_t Bs[NTILE * KP2];                 // 32 KB
    __shared__ float  s_sum[O_], s_sq[O_];

    int tid = threadIdx.x;
    int colBase = blockIdx.x * NTILE;
    s_sum[tid] = 0.0f; s_sq[tid] = 0.0f;

    // ---- stage: 4 threads per column, BN1+ReLU on the fly ----
    int col = tid >> 2, r = tid & 3;
    int gcol = colBase + col;
    const half_t* ycol = y1 + (size_t)gcol * O_ + r * 64;
    half_t*       dst  = Bs + (size_t)col * KP2 + r * 64;
    #pragma unroll
    for (int j = 0; j < 8; ++j) {
        v8h hv = *(const v8h*)(ycol + j * 8);
        v8h ov;
        #pragma unroll
        for (int e = 0; e < 8; ++e) {
            int k = r * 64 + j * 8 + e;
            float v = (float)hv[e];
            v = (v - mu1[k]) * rs1[k] * g1[k] + b1[k];
            ov[e] = (half_t)(v > 0.0f ? v : 0.0f);
        }
        *(v8h*)(dst + j * 8) = ov;
    }
    __syncthreads();

    int wave = tid >> 5, lane = tid & 31;
    int m0 = wave * 32;
    int n  = lane & 15, hi = lane >> 4;
    v8f acc[8];
    #pragma unroll
    for (int q = 0; q < 8; ++q) acc[q] = (v8f){};
    const half_t* arow0 = W2h + (size_t)(m0 + n) * KP2 + hi * 8;
    const half_t* arow1 = arow0 + (size_t)16 * KP2;

    #pragma unroll
    for (int k0 = 0; k0 < KP2; k0 += 32) {
        v16h a0 = cat8(*(const v8h*)(arow0 + k0), *(const v8h*)(arow0 + k0 + 16));
        v16h a1 = cat8(*(const v8h*)(arow1 + k0), *(const v8h*)(arow1 + k0 + 16));
        #pragma unroll
        for (int nt = 0; nt < 4; ++nt) {
            const half_t* bcol = Bs + (size_t)(nt * 16 + n) * KP2 + hi * 8 + k0;
            v16h bb = cat8(*(const v8h*)(bcol), *(const v8h*)(bcol + 16));
            acc[nt]     = __builtin_amdgcn_wmma_f32_16x16x32_f16(false, a0, false, bb, (short)0, acc[nt],     false, false);
            acc[4 + nt] = __builtin_amdgcn_wmma_f32_16x16x32_f16(false, a1, false, bb, (short)0, acc[4 + nt], false, false);
        }
    }

    #pragma unroll
    for (int nt = 0; nt < 4; ++nt) {
        v8h pk0, pk1;
        #pragma unroll
        for (int rr = 0; rr < 8; ++rr) {
            float v0 = acc[nt][rr], v1 = acc[4 + nt][rr];
            int o0 = m0 + hi * 8 + rr, o1 = o0 + 16;
            atomicAdd(&s_sum[o0], v0); atomicAdd(&s_sq[o0], v0 * v0);
            atomicAdd(&s_sum[o1], v1); atomicAdd(&s_sq[o1], v1 * v1);
            pk0[rr] = (half_t)v0; pk1[rr] = (half_t)v1;
        }
        half_t* yrow = y2 + (size_t)(colBase + nt * 16 + n) * O_;
        *(v8h*)(yrow + m0 + hi * 8)      = pk0;
        *(v8h*)(yrow + m0 + 16 + hi * 8) = pk1;
    }

    __syncthreads();
    atomicAdd(&gsum[tid], s_sum[tid]);
    atomicAdd(&gsq[tid],  s_sq[tid]);
}

// ---------------- K6: BN2 + ReLU + max over NS -> out (B,256,P) ----------------
__global__ __launch_bounds__(256)
void maxout_kernel(const half_t* __restrict__ y2, const float* __restrict__ mu2,
                   const float* __restrict__ rs2, const float* __restrict__ g2,
                   const float* __restrict__ b2, float* __restrict__ out) {
    int bp = blockIdx.x;                               // b*N + p
    int o  = threadIdx.x;
    int b  = bp >> 10, p = bp & (N_ - 1);
    float mu = mu2[o], rs = rs2[o], g = g2[o], bb = b2[o];
    float m = 0.0f;                                    // post-ReLU max >= 0
    const half_t* base = y2 + ((size_t)bp * NS_) * O_ + o;
    #pragma unroll 4
    for (int s = 0; s < NS_; ++s) {
        float v = (float)base[(size_t)s * O_];
        v = (v - mu) * rs * g + bb;
        v = v > 0.0f ? v : 0.0f;
        m = v > m ? v : m;
    }
    out[((size_t)(b * O_ + o)) * N_ + p] = m;
}

// ---------------- host launch ----------------
extern "C" void kernel_launch(void* const* d_in, const int* in_sizes, int n_in,
                              void* d_out, int out_size, void* d_ws, size_t ws_size,
                              hipStream_t stream) {
    const float* xyz   = (const float*)d_in[0];
    const float* feats = (const float*)d_in[1];
    const float* rot   = (const float*)d_in[2];
    const float* W1    = (const float*)d_in[3];
    const float* g1    = (const float*)d_in[4];
    const float* b1    = (const float*)d_in[5];
    const float* W2    = (const float*)d_in[6];
    const float* g2    = (const float*)d_in[7];
    const float* b2    = (const float*)d_in[8];
    float* out = (float*)d_out;

    char* ws = (char*)d_ws;
    // workspace layout (aligned slabs)
    int*    idxbuf = (int*)    (ws + 0);                            //   1 MB
    half_t* featsT = (half_t*) (ws + 1048576);                      //   4 MB
    half_t* W1h    = (half_t*) (ws + 5242880);                      // 144 KB
    half_t* W2h    = (half_t*) (ws + 5390336);                      // 128 KB
    float*  stats  = (float*)  (ws + 5521408);                      //   8 KB (2048 f32)
    half_t* y1     = (half_t*) (ws + 5529600);                      // 128 MB
    half_t* y2     = (half_t*) (ws + 139747328);                    // 128 MB

    float* st1 = stats;         // sum1 sq1 mu1 rs1
    float* st2 = stats + 1024;  // sum2 sq2 mu2 rs2

    zero_stats_kernel<<<2, 1024, 0, stream>>>(stats);
    prep_feats_kernel<<<(B_ * N_ * C_) / 256, 256, 0, stream>>>(feats, featsT);
    prep_w1_kernel<<<(O_ * KP1 + 255) / 256, 256, 0, stream>>>(W1, W1h);
    prep_w2_kernel<<<(O_ * KP2 + 255) / 256, 256, 0, stream>>>(W2, W2h);
    cyl_query_kernel<<<(B_ * N_) / 8, 256, 0, stream>>>(xyz, rot, idxbuf);
    gemm1_kernel<<<COLS / NTILE, 256, 0, stream>>>(idxbuf, xyz, rot, featsT, W1h, y1,
                                                   st1, st1 + 256);
    finalize_stats_kernel<<<1, 256, 0, stream>>>(st1);
    gemm2_kernel<<<COLS / NTILE, 256, 0, stream>>>(y1, st1 + 512, st1 + 768, g1, b1,
                                                   W2h, y2, st2, st2 + 256);
    finalize_stats_kernel<<<1, 256, 0, stream>>>(st2);
    maxout_kernel<<<B_ * N_, 256, 0, stream>>>(y2, st2 + 512, st2 + 768, g2, b2, out);
}
